// SparseMambaAttax_68985764708575
// MI455X (gfx1250) — compile-verified, matrix operand order checked by reading, not run on hardware
//
#include <hip/hip_runtime.h>
#include <hip/hip_bf16.h>
#include <cstdint>
#include <cstddef>

// ---------------------------------------------------------------------------
// SparseMambaAttax forward for MI455X (gfx1250, wave32, WMMA bf16 path)
// ---------------------------------------------------------------------------
static constexpr int SEQ      = 2048;
static constexpr int DM       = 1024;   // D_MODEL == D_INNER
static constexpr int IN_DIM   = 2208;   // 2*1024 + 2*64 + 32
static constexpr int CONV_CH  = 1152;   // 1024 + 2*64
static constexpr int M_NH     = 32;
static constexpr int M_HD     = 32;
static constexpr int M_DS     = 64;
static constexpr int NH       = 16;     // attention heads
static constexpr int VD       = 64;
static constexpr int RD       = 32;     // rope dim
static constexpr int HD       = 96;     // VD + RD
static constexpr int TOPK     = 64;

typedef __attribute__((ext_vector_type(16))) __bf16 v16bf;
typedef __attribute__((ext_vector_type(8)))  float  v8f;
typedef __attribute__((ext_vector_type(4)))  int    v4i;

// Async global->LDS path (CDNA5); guarded so compile never regresses.
#if defined(__has_builtin)
# if __has_builtin(__builtin_amdgcn_global_load_async_to_lds_b128)
#  define USE_ASYNC_LDS 1
# endif
#endif
#ifndef USE_ASYNC_LDS
# define USE_ASYNC_LDS 0
#endif

#define AS_GLOBAL __attribute__((address_space(1)))
#define AS_LDS    __attribute__((address_space(3)))

#if USE_ASYNC_LDS
__device__ __forceinline__ void async_copy16(const void* g, void* l) {
    // global_load_async_to_lds_b128: (v4i AS1* src, v4i AS3* dst, imm off, imm cpol)
    __builtin_amdgcn_global_load_async_to_lds_b128(
        (AS_GLOBAL v4i*)g, (AS_LDS v4i*)l, 0, 0);
}
#endif

__device__ __forceinline__ void wait_async_lds() {
#if USE_ASYNC_LDS
# if __has_builtin(__builtin_amdgcn_s_wait_asynccnt)
    __builtin_amdgcn_s_wait_asynccnt(0);
# else
    asm volatile("s_wait_asynccnt 0x0" ::: "memory");
# endif
#endif
}

// ---------------------------------------------------------------------------
// f32 -> bf16 convert
// ---------------------------------------------------------------------------
__global__ void k_cvt_bf16(const float* __restrict__ src,
                           __hip_bfloat16* __restrict__ dst, int n) {
    int i = blockIdx.x * blockDim.x + threadIdx.x;
    if (i < n) dst[i] = __float2bfloat16(src[i]);
}

// ---------------------------------------------------------------------------
// WMMA GEMM: C[M,N] = scale * A[M,K] * B[N,K]^T   (A,B bf16 row-major, C f32)
// Optionally also writes bf16 copy Cb. M % 128 == 0, N % 32 == 0, K % 32 == 0.
// Block = 256 threads = 8 waves; block tile 128x32.
// Each wave owns two 16x16 M-tiles sharing one B fragment: 2 WMMAs / K-step.
// A/B tiles staged through LDS via global_load_async_to_lds_b128 (ASYNCcnt)
// when available, else vectorized b128 loads.
// ---------------------------------------------------------------------------
__global__ __launch_bounds__(256)
void k_gemm_bf16(const __hip_bfloat16* __restrict__ A,
                 const __hip_bfloat16* __restrict__ B,
                 float* __restrict__ C,
                 __hip_bfloat16* __restrict__ Cb,
                 int M, int N, int K, float scale) {
    __shared__ __align__(16) __hip_bfloat16 As[128][32];
    __shared__ __align__(16) __hip_bfloat16 Bs[32][32];

    const int tid   = threadIdx.x;
    const int lane  = tid & 31;
    const int wave  = tid >> 5;        // 0..7
    const int wm    = wave & 3;        // wave M-row pair within block tile
    const int wn    = wave >> 2;       // wave N-col within block tile
    const int m0    = blockIdx.y * 128;
    const int n0    = blockIdx.x * 32;
    const int row   = lane & 15;       // M (or N) index within 16x16 tile
    const int khalf = lane >> 4;       // K sub-half select per ISA layout

    v8f acc0 = {};
    v8f acc1 = {};

    for (int k0 = 0; k0 < K; k0 += 32) {
        // ---- stage A (128x32 bf16 = 512 x 16B) and B (128 x 16B) ----
        // Each 16B chunk: row r, chunk c -> elements [k0 + c*8, +8)
        {
            const __hip_bfloat16* Abase = A + (size_t)m0 * K + k0;
            const __hip_bfloat16* Bbase = B + (size_t)n0 * K + k0;
#if USE_ASYNC_LDS
            #pragma unroll
            for (int i = 0; i < 2; ++i) {
                int e = tid + i * 256;             // 0..511
                int r = e >> 2, c = e & 3;
                async_copy16(Abase + (size_t)r * K + c * 8, &As[r][c * 8]);
            }
            if (tid < 128) {
                int r = tid >> 2, c = tid & 3;
                async_copy16(Bbase + (size_t)r * K + c * 8, &Bs[r][c * 8]);
            }
            wait_async_lds();
#else
            #pragma unroll
            for (int i = 0; i < 2; ++i) {
                int e = tid + i * 256;
                int r = e >> 2, c = e & 3;
                *(uint4*)&As[r][c * 8] =
                    *(const uint4*)(Abase + (size_t)r * K + c * 8);
            }
            if (tid < 128) {
                int r = tid >> 2, c = tid & 3;
                *(uint4*)&Bs[r][c * 8] =
                    *(const uint4*)(Bbase + (size_t)r * K + c * 8);
            }
#endif
            if (k0 + 32 < K) {
                __builtin_prefetch(Abase + (size_t)(tid >> 1) * K + 32, 0, 1);
                __builtin_prefetch(Bbase + (size_t)(tid >> 3) * K + 32, 0, 1);
            }
        }
        __syncthreads();

        // ---- fragments per CDNA5 16-bit layout ----
        // lane 0-15: rows 0-15, K = {0..7,16..23}; lane 16-31: K = {8..15,24..31}
        union { v16bf v; uint32_t u[8]; } a0, a1, bb;
        const uint32_t* Ar0 = (const uint32_t*)&As[wm * 16 + row][0];
        const uint32_t* Ar1 = (const uint32_t*)&As[64 + wm * 16 + row][0];
        const uint32_t* Br  = (const uint32_t*)&Bs[wn * 16 + row][0];
        #pragma unroll
        for (int v = 0; v < 8; ++v) {
            int kb = (v & 3) + khalf * 4 + (v >> 2) * 8;   // dword idx in row
            a0.u[v] = Ar0[kb];
            a1.u[v] = Ar1[kb];
            bb.u[v] = Br[kb];
        }
        acc0 = __builtin_amdgcn_wmma_f32_16x16x32_bf16(
                   false, a0.v, false, bb.v, (short)0, acc0, false, false);
        acc1 = __builtin_amdgcn_wmma_f32_16x16x32_bf16(
                   false, a1.v, false, bb.v, (short)0, acc1, false, false);
        __syncthreads();
    }

    // C/D layout: lane&15 = N col; lane>>4 selects M rows {0..7} vs {8..15}.
    const int cn  = n0 + wn * 16 + (lane & 15);
    const int mb0 = m0 + wm * 16 + ((lane >> 4) * 8);
    const int mb1 = mb0 + 64;
    #pragma unroll
    for (int r = 0; r < 8; ++r) {
        float v0 = acc0[r] * scale;
        float v1 = acc1[r] * scale;
        size_t o0 = (size_t)(mb0 + r) * N + cn;
        size_t o1 = (size_t)(mb1 + r) * N + cn;
        C[o0] = v0;
        C[o1] = v1;
        if (Cb) { Cb[o0] = __float2bfloat16(v0); Cb[o1] = __float2bfloat16(v1); }
    }
}

// ---------------------------------------------------------------------------
// Depthwise causal conv (4 taps) over xBC channels + SiLU
// ---------------------------------------------------------------------------
__global__ void k_conv_silu(const float* __restrict__ zxbcdt,
                            const float* __restrict__ conv_w,
                            const float* __restrict__ conv_b,
                            float* __restrict__ xbc_act) {
    int g = blockIdx.x * blockDim.x + threadIdx.x;
    if (g >= SEQ * CONV_CH) return;
    int l = g / CONV_CH, c = g % CONV_CH;
    float acc = conv_b[c];
    #pragma unroll
    for (int k = 0; k < 4; ++k) {
        int lr = l - 3 + k;
        if (lr >= 0)
            acc += zxbcdt[(size_t)lr * IN_DIM + DM + c] * conv_w[c * 4 + k];
    }
    xbc_act[g] = acc / (1.0f + __expf(-acc));   // silu
}

// ---------------------------------------------------------------------------
// dt = softplus(dt_raw + bias);  dA = exp(-exp(A_log)*dt)
// ---------------------------------------------------------------------------
__global__ void k_dt(const float* __restrict__ zxbcdt,
                     const float* __restrict__ dt_bias,
                     const float* __restrict__ A_log,
                     float* __restrict__ dts, float* __restrict__ dAs) {
    int g = blockIdx.x * blockDim.x + threadIdx.x;
    if (g >= SEQ * M_NH) return;
    int l = g / M_NH, h = g % M_NH;
    float v  = zxbcdt[(size_t)l * IN_DIM + (IN_DIM - M_NH) + h] + dt_bias[h];
    float dt = (v > 20.f) ? v : log1pf(__expf(v));
    dts[g] = dt;
    dAs[g] = __expf(-__expf(A_log[h]) * dt);
}

// ---------------------------------------------------------------------------
// Mamba2 selective scan. One block per head (32 blocks, 512 threads).
// Thread (p = tid>>4, ng = tid&15) owns 4 state elements n = ng*4..ng*4+3.
// Reduction over n via wave32 __shfl_xor within width-16 groups.
// ---------------------------------------------------------------------------
__global__ __launch_bounds__(512)
void k_scan(const float* __restrict__ xbc,
            const float* __restrict__ dts,
            const float* __restrict__ dAs,
            const float* __restrict__ Dp,
            float* __restrict__ y) {
    const int h   = blockIdx.x;
    const int tid = threadIdx.x;
    const int p   = tid >> 4;
    const int ng  = tid & 15;
    float st[4] = {0.f, 0.f, 0.f, 0.f};
    __shared__ float Bs[M_DS], Cs[M_DS], xsh[M_HD], sc[2];
    const float Dh = Dp[h];

    for (int t = 0; t < SEQ; ++t) {
        const float* rowp = xbc + (size_t)t * CONV_CH;
        if (tid < 64)        Bs[tid]        = rowp[DM + tid];
        else if (tid < 128)  Cs[tid - 64]   = rowp[DM + M_DS + (tid - 64)];
        else if (tid < 160)  xsh[tid - 128] = rowp[h * M_HD + (tid - 128)];
        else if (tid == 160) { sc[0] = dts[t * M_NH + h]; sc[1] = dAs[t * M_NH + h]; }
        __syncthreads();

        const float dt = sc[0], dA = sc[1];
        const float xv = xsh[p];
        const float dbx = dt * xv;
        float part = 0.f;
        #pragma unroll
        for (int j = 0; j < 4; ++j) {
            int n = ng * 4 + j;
            st[j] = st[j] * dA + dbx * Bs[n];
            part += st[j] * Cs[n];
        }
        #pragma unroll
        for (int off = 8; off >= 1; off >>= 1)
            part += __shfl_xor(part, off, 16);
        if (ng == 0)
            y[(size_t)t * DM + h * M_HD + p] = part + Dh * xv;
        __syncthreads();
    }
}

// ---------------------------------------------------------------------------
// y * silu(z), RMS-norm over 1024, * norm_w -> bf16
// ---------------------------------------------------------------------------
__global__ __launch_bounds__(256)
void k_gatenorm(const float* __restrict__ y, const float* __restrict__ zx,
                const float* __restrict__ norm_w,
                __hip_bfloat16* __restrict__ out) {
    const int l = blockIdx.x, tid = threadIdx.x;
    __shared__ float red[256];
    float vals[4]; float ss = 0.f;
    #pragma unroll
    for (int i = 0; i < 4; ++i) {
        int c = tid + i * 256;
        float z = zx[(size_t)l * IN_DIM + c];
        float v = y[(size_t)l * DM + c] * (z / (1.f + __expf(-z)));
        vals[i] = v; ss += v * v;
    }
    red[tid] = ss; __syncthreads();
    for (int off = 128; off > 0; off >>= 1) {
        if (tid < off) red[tid] += red[tid + off];
        __syncthreads();
    }
    float inv = rsqrtf(red[0] / (float)DM + 1e-5f);
    #pragma unroll
    for (int i = 0; i < 4; ++i) {
        int c = tid + i * 256;
        out[(size_t)l * DM + c] = __float2bfloat16(vals[i] * inv * norm_w[c]);
    }
}

// ---------------------------------------------------------------------------
// Causal top-64 per row (iterative argmax, ties -> lowest index like jax)
// ---------------------------------------------------------------------------
__global__ __launch_bounds__(256)
void k_topk(const float* __restrict__ scores, int* __restrict__ idxout) {
    __shared__ float sv[SEQ];
    __shared__ uint32_t picked[SEQ / 32];
    __shared__ float rv[256];
    __shared__ int   ri[256];
    const int l = blockIdx.x, tid = threadIdx.x;
    for (int i = tid; i < SEQ; i += 256)
        sv[i] = (i <= l) ? scores[(size_t)l * SEQ + i] : -INFINITY;
    for (int i = tid; i < SEQ / 32; i += 256) picked[i] = 0u;
    __syncthreads();
    for (int s = 0; s < TOPK; ++s) {
        float bv = -INFINITY; int bi = SEQ;
        for (int i = tid; i < SEQ; i += 256) {
            if (picked[i >> 5] & (1u << (i & 31))) continue;
            float v = sv[i];
            if (v > bv || (v == bv && i < bi)) { bv = v; bi = i; }
        }
        rv[tid] = bv; ri[tid] = bi;
        __syncthreads();
        for (int off = 128; off > 0; off >>= 1) {
            if (tid < off) {
                float ov = rv[tid + off]; int oi = ri[tid + off];
                if (ov > rv[tid] || (ov == rv[tid] && oi < ri[tid])) {
                    rv[tid] = ov; ri[tid] = oi;
                }
            }
            __syncthreads();
        }
        if (tid == 0) {
            int w = ri[0];
            idxout[(size_t)l * TOPK + s] = w;
            picked[w >> 5] |= 1u << (w & 31);
        }
        __syncthreads();
    }
}

// ---------------------------------------------------------------------------
// Assemble q_final / k_final (content + rotated rope) and v as bf16
// ---------------------------------------------------------------------------
__global__ __launch_bounds__(256)
void k_buildqk(const float* __restrict__ qc, const float* __restrict__ qr,
               const float* __restrict__ kv, const float* __restrict__ krs,
               __hip_bfloat16* __restrict__ qf, __hip_bfloat16* __restrict__ kf,
               __hip_bfloat16* __restrict__ vb) {
    const int l = blockIdx.x, tid = threadIdx.x;
    __shared__ float sn[16], cs[16];
    if (tid < 16) {
        // inv = 10000^(-tid/16);  ln(10000)/16 = 0.57564627324851f
        float ang = (float)l * __expf(-0.57564627324851f * (float)tid);
        sn[tid] = __sinf(ang); cs[tid] = __cosf(ang);
    }
    __syncthreads();
    for (int e = tid; e < NH * HD; e += 256) {
        int h = e / HD, d = e % HD;
        float qv, kvv;
        if (d < VD) {
            qv  = qc[(size_t)l * (NH * VD) + h * VD + d];
            kvv = kv[(size_t)l * (2 * NH * VD) + h * VD + d];
        } else {
            int i  = d - VD;
            int ii = i & 15;
            float qx1 = qr[(size_t)l * (NH * RD) + h * RD + ii];
            float qx2 = qr[(size_t)l * (NH * RD) + h * RD + 16 + ii];
            float kx1 = krs[(size_t)l * RD + ii];
            float kx2 = krs[(size_t)l * RD + 16 + ii];
            if (i < 16) { qv = qx1 * cs[ii] - qx2 * sn[ii]; kvv = kx1 * cs[ii] - kx2 * sn[ii]; }
            else        { qv = qx1 * sn[ii] + qx2 * cs[ii]; kvv = kx1 * sn[ii] + kx2 * cs[ii]; }
        }
        qf[(size_t)l * (NH * HD) + e] = __float2bfloat16(qv);
        kf[(size_t)l * (NH * HD) + e] = __float2bfloat16(kvv);
    }
    for (int e = tid; e < NH * VD; e += 256)
        vb[(size_t)l * (NH * VD) + e] =
            __float2bfloat16(kv[(size_t)l * (2 * NH * VD) + NH * VD + e]);
}

// ---------------------------------------------------------------------------
// Sparse gather attention: one block per query row
// ---------------------------------------------------------------------------
__global__ __launch_bounds__(128)
void k_attn(const __hip_bfloat16* __restrict__ qf,
            const __hip_bfloat16* __restrict__ kf,
            const __hip_bfloat16* __restrict__ vb,
            const int* __restrict__ idx,
            __hip_bfloat16* __restrict__ attn_b) {
    __shared__ int   id[TOPK];
    __shared__ float qh[HD];
    __shared__ float lg[TOPK];
    __shared__ float inv_s;
    const int l = blockIdx.x, tid = threadIdx.x;
    if (tid < TOPK) id[tid] = idx[(size_t)l * TOPK + tid];
    __syncthreads();
    const float scale = rsqrtf((float)HD);
    for (int h = 0; h < NH; ++h) {
        if (tid < HD) qh[tid] = __bfloat162float(qf[((size_t)l * NH + h) * HD + tid]);
        __syncthreads();
        if (tid < TOPK) {
            const __hip_bfloat16* kr = kf + ((size_t)id[tid] * NH + h) * HD;
            float s = 0.f;
            #pragma unroll 8
            for (int d = 0; d < HD; ++d) s += qh[d] * __bfloat162float(kr[d]);
            lg[tid] = s * scale;
        }
        __syncthreads();
        if (tid == 0) {
            float mx = lg[0];
            for (int j = 1; j < TOPK; ++j) mx = fmaxf(mx, lg[j]);
            float sm = 0.f;
            for (int j = 0; j < TOPK; ++j) { float e = __expf(lg[j] - mx); lg[j] = e; sm += e; }
            inv_s = 1.0f / sm;
        }
        __syncthreads();
        if (tid < VD) {
            float acc = 0.f;
            #pragma unroll 8
            for (int j = 0; j < TOPK; ++j)
                acc += lg[j] * __bfloat162float(vb[((size_t)id[j] * NH + h) * VD + tid]);
            attn_b[(size_t)l * (NH * VD) + h * VD + tid] = __float2bfloat16(acc * inv_s);
        }
        __syncthreads();
    }
}

// ---------------------------------------------------------------------------
// Host side
// ---------------------------------------------------------------------------
static inline void* bump(char* base, size_t& off, size_t bytes) {
    void* r = base + off;
    off = (off + bytes + 255) & ~(size_t)255;
    return r;
}

static void gemm(const __hip_bfloat16* A, const __hip_bfloat16* B,
                 float* C, __hip_bfloat16* Cb,
                 int M, int N, int K, float scale, hipStream_t s) {
    dim3 grid(N / 32, M / 128);
    k_gemm_bf16<<<grid, 256, 0, s>>>(A, B, C, Cb, M, N, K, scale);
}

static void cvt(const float* src, __hip_bfloat16* dst, int n, hipStream_t s) {
    k_cvt_bf16<<<(n + 255) / 256, 256, 0, s>>>(src, dst, n);
}

extern "C" void kernel_launch(void* const* d_in, const int* in_sizes, int n_in,
                              void* d_out, int out_size, void* d_ws, size_t ws_size,
                              hipStream_t stream) {
    (void)in_sizes; (void)n_in; (void)out_size; (void)ws_size;

    const float* x         = (const float*)d_in[0];
    const float* idx_in_w  = (const float*)d_in[1];
    const float* idx_conv_w= (const float*)d_in[2];
    const float* idx_conv_b= (const float*)d_in[3];
    const float* idx_dt_b  = (const float*)d_in[4];
    const float* idx_A_log = (const float*)d_in[5];
    const float* idx_D     = (const float*)d_in[6];
    const float* idx_norm_w= (const float*)d_in[7];
    const float* idx_out_w = (const float*)d_in[8];
    const float* idx_q_w   = (const float*)d_in[9];
    const float* idx_k_w   = (const float*)d_in[10];
    const float* q_down_w  = (const float*)d_in[11];
    const float* q_up_w    = (const float*)d_in[12];
    const float* q_rope_w  = (const float*)d_in[13];
    const float* kv_down_w = (const float*)d_in[14];
    const float* kv_up_w   = (const float*)d_in[15];
    const float* k_rope_w  = (const float*)d_in[16];
    const float* out_w     = (const float*)d_in[17];
    float* out = (float*)d_out;

    char* ws = (char*)d_ws;
    size_t off = 0;
    typedef __hip_bfloat16 bf;

    // bf16 copies of activations-input and weights
    bf* xb        = (bf*)bump(ws, off, (size_t)SEQ * DM * 2);
    bf* w_in_b    = (bf*)bump(ws, off, (size_t)IN_DIM * DM * 2);
    bf* w_out_b   = (bf*)bump(ws, off, (size_t)DM * DM * 2);
    bf* w_qidx_b  = (bf*)bump(ws, off, (size_t)64 * DM * 2);
    bf* w_kidx_b  = (bf*)bump(ws, off, (size_t)64 * DM * 2);
    bf* w_qdn_b   = (bf*)bump(ws, off, (size_t)128 * DM * 2);
    bf* w_qup_b   = (bf*)bump(ws, off, (size_t)1024 * 128 * 2);
    bf* w_qrp_b   = (bf*)bump(ws, off, (size_t)512 * 128 * 2);
    bf* w_kvd_b   = (bf*)bump(ws, off, (size_t)128 * DM * 2);
    bf* w_kvu_b   = (bf*)bump(ws, off, (size_t)2048 * 128 * 2);
    bf* w_krp_b   = (bf*)bump(ws, off, (size_t)RD * DM * 2);
    bf* w_ow_b    = (bf*)bump(ws, off, (size_t)DM * DM * 2);

    // f32 intermediates
    float* zxbcdt  = (float*)bump(ws, off, (size_t)SEQ * IN_DIM * 4);
    float* xbc_act = (float*)bump(ws, off, (size_t)SEQ * CONV_CH * 4);
    float* dts     = (float*)bump(ws, off, (size_t)SEQ * M_NH * 4);
    float* dAs     = (float*)bump(ws, off, (size_t)SEQ * M_NH * 4);
    float* y_raw   = (float*)bump(ws, off, (size_t)SEQ * DM * 4);
    bf*    ymix_b  = (bf*)bump(ws, off, (size_t)SEQ * DM * 2);
    float* x_mamba = (float*)bump(ws, off, (size_t)SEQ * DM * 4);
    bf*    x_mam_b = (bf*)bump(ws, off, (size_t)SEQ * DM * 2);
    float* q_idx_f = (float*)bump(ws, off, (size_t)SEQ * 64 * 4);
    bf*    q_idx_b = (bf*)bump(ws, off, (size_t)SEQ * 64 * 2);
    float* k_idx_f = (float*)bump(ws, off, (size_t)SEQ * 64 * 4);
    bf*    k_idx_b = (bf*)bump(ws, off, (size_t)SEQ * 64 * 2);
    float* scores  = (float*)bump(ws, off, (size_t)SEQ * SEQ * 4);
    int*   indices = (int*)bump(ws, off, (size_t)SEQ * TOPK * 4);
    float* c_q_f   = (float*)bump(ws, off, (size_t)SEQ * 128 * 4);
    bf*    c_q_b   = (bf*)bump(ws, off, (size_t)SEQ * 128 * 2);
    float* q_cont  = (float*)bump(ws, off, (size_t)SEQ * 1024 * 4);
    float* q_rope  = (float*)bump(ws, off, (size_t)SEQ * 512 * 4);
    float* c_kv_f  = (float*)bump(ws, off, (size_t)SEQ * 128 * 4);
    bf*    c_kv_b  = (bf*)bump(ws, off, (size_t)SEQ * 128 * 2);
    float* kvf     = (float*)bump(ws, off, (size_t)SEQ * 2048 * 4);
    float* krs     = (float*)bump(ws, off, (size_t)SEQ * RD * 4);
    bf*    qf_b    = (bf*)bump(ws, off, (size_t)SEQ * NH * HD * 2);
    bf*    kf_b    = (bf*)bump(ws, off, (size_t)SEQ * NH * HD * 2);
    bf*    vb_b    = (bf*)bump(ws, off, (size_t)SEQ * NH * VD * 2);
    bf*    attn_b  = (bf*)bump(ws, off, (size_t)SEQ * DM * 2);

    // ---- stage 0: bf16 conversions ----
    cvt(x,         xb,       SEQ * DM,       stream);
    cvt(idx_in_w,  w_in_b,   IN_DIM * DM,    stream);
    cvt(idx_out_w, w_out_b,  DM * DM,        stream);
    cvt(idx_q_w,   w_qidx_b, 64 * DM,        stream);
    cvt(idx_k_w,   w_kidx_b, 64 * DM,        stream);
    cvt(q_down_w,  w_qdn_b,  128 * DM,       stream);
    cvt(q_up_w,    w_qup_b,  1024 * 128,     stream);
    cvt(q_rope_w,  w_qrp_b,  512 * 128,      stream);
    cvt(kv_down_w, w_kvd_b,  128 * DM,       stream);
    cvt(kv_up_w,   w_kvu_b,  2048 * 128,     stream);
    cvt(k_rope_w,  w_krp_b,  RD * DM,        stream);
    cvt(out_w,     w_ow_b,   DM * DM,        stream);

    // ---- Mamba2 branch ----
    gemm(xb, w_in_b, zxbcdt, nullptr, SEQ, IN_DIM, DM, 1.0f, stream);
    k_conv_silu<<<(SEQ * CONV_CH + 255) / 256, 256, 0, stream>>>(
        zxbcdt, idx_conv_w, idx_conv_b, xbc_act);
    k_dt<<<(SEQ * M_NH + 255) / 256, 256, 0, stream>>>(
        zxbcdt, idx_dt_b, idx_A_log, dts, dAs);
    k_scan<<<M_NH, 512, 0, stream>>>(xbc_act, dts, dAs, idx_D, y_raw);
    k_gatenorm<<<SEQ, 256, 0, stream>>>(y_raw, zxbcdt, idx_norm_w, ymix_b);
    gemm(ymix_b, w_out_b, x_mamba, x_mam_b, SEQ, DM, DM, 1.0f, stream);

    // ---- index scores + top-k ----
    gemm(x_mam_b, w_qidx_b, q_idx_f, q_idx_b, SEQ, 64, DM, 1.0f, stream);
    gemm(xb,      w_kidx_b, k_idx_f, k_idx_b, SEQ, 64, DM, 1.0f, stream);
    gemm(q_idx_b, k_idx_b,  scores,  nullptr, SEQ, SEQ, 64, 0.125f, stream);
    k_topk<<<SEQ, 256, 0, stream>>>(scores, indices);

    // ---- low-rank Q/KV projections ----
    gemm(xb,     w_qdn_b, c_q_f,  c_q_b,   SEQ, 128,  DM,  1.0f, stream);
    gemm(c_q_b,  w_qup_b, q_cont, nullptr, SEQ, 1024, 128, 1.0f, stream);
    gemm(c_q_b,  w_qrp_b, q_rope, nullptr, SEQ, 512,  128, 1.0f, stream);
    gemm(xb,     w_kvd_b, c_kv_f, c_kv_b,  SEQ, 128,  DM,  1.0f, stream);
    gemm(c_kv_b, w_kvu_b, kvf,    nullptr, SEQ, 2048, 128, 1.0f, stream);
    gemm(xb,     w_krp_b, krs,    nullptr, SEQ, RD,   DM,  1.0f, stream);

    // ---- RoPE assembly + sparse attention ----
    k_buildqk<<<SEQ, 256, 0, stream>>>(q_cont, q_rope, kvf, krs, qf_b, kf_b, vb_b);
    k_attn<<<SEQ, 128, 0, stream>>>(qf_b, kf_b, vb_b, indices, attn_b);

    // ---- output projection ----
    gemm(attn_b, w_ow_b, out, nullptr, SEQ, DM, DM, 1.0f, stream);
}
